// VIB_14267881357871
// MI455X (gfx1250) — compile-verified, matrix operand order checked by reading, not run on hardware
//
#include <hip/hip_runtime.h>
#include <stdint.h>

typedef __attribute__((ext_vector_type(16))) __bf16 v16bf;
typedef __attribute__((ext_vector_type(8)))  float  v8f;

#define D_IN  784
#define DZ    128
#define NCLS  10
#define KC    64          // K-chunk staged in LDS
#define KPAD  832         // 13 * 64, zero-padded past 784
#define NCH   (KPAD / KC) // 13 chunks
#define MT    64          // rows per block
#define SRS   72          // LDS row stride in bf16 halves (144B = 16*9: b128-aligned, staggered banks)
#define SHS   132         // f32 h-tile row stride (conflict-free strided epilogue reads)
#define KPROP 64

// bf16 arena: sxh(64*72) + sxl(64*72) + swh(128*72) + swl(128*72) = 27648 halves = 55296 B
#define SX_H   (MT * SRS)
#define SW_H   (DZ * SRS)
#define ARENA_H (2 * SX_H + 2 * SW_H)
// epilogue f32 h-tile aliases arena: 64*132*4 = 33792 B  (< 55296 B, fits)

union H4   { __bf16 h[4]; uint2 u; };
union FRAG { uint4 q[2]; v16bf v; };

__device__ __forceinline__ v16bf ldfrag(const __bf16* rowp, int off) {
    FRAG f;
    f.q[0] = *(const uint4*)(rowp + off);        // K = off .. off+7
    f.q[1] = *(const uint4*)(rowp + off + 16);   // K = off+16 .. off+23
    return f.v;
}

__device__ __forceinline__ float redg4(float v) {
    v += __shfl_xor(v, 1, 32);
    v += __shfl_xor(v, 2, 32);
    return v;
}

__device__ __forceinline__ uint32_t hmix(uint32_t x) {
    x ^= x >> 16; x *= 0x7feb352du;
    x ^= x >> 15; x *= 0x846ca68bu;
    x ^= x >> 16;
    return x;
}

__device__ __forceinline__ float u01(uint32_t h) {
    return ((float)(h >> 8) + 0.5f) * (1.0f / 16777216.0f);   // (0,1)
}

__device__ __forceinline__ float gauss(uint32_t h1, uint32_t h2) {
    float u1 = u01(h1), u2 = u01(h2);
    return sqrtf(-2.0f * logf(u1)) * cosf(6.28318530718f * u2);
}

__global__ __launch_bounds__(256) void vib_vmf_kernel(
    const float* __restrict__ x,    const float* __restrict__ Wenc,
    const float* __restrict__ benc, const float* __restrict__ Wvar,
    const float* __restrict__ bvar, const float* __restrict__ Wdec,
    const float* __restrict__ bdec, float* __restrict__ out, int B)
{
    __shared__ __align__(16) unsigned short arena[ARENA_H];
    __shared__ float c_benc[DZ];
    __shared__ float c_wvar[DZ];
    __shared__ float c_wdec[DZ * NCLS];
    __shared__ float c_bdec[NCLS];
    __shared__ float c_bvar;

    __bf16* sxh = (__bf16*)arena;          // A tile, hi halves (row-major)
    __bf16* sxl = sxh + SX_H;              // A tile, lo halves
    __bf16* swh = sxl + SX_H;              // B tile, hi halves (column-major: [n][k])
    __bf16* swl = swh + SW_H;              // B tile, lo halves
    float*  sh  = (float*)arena;           // epilogue h tile (aliases GEMM arena)

    const int tid     = threadIdx.x;
    const int rowBase = blockIdx.x * MT;

    // ---- preload small constants into LDS ----
    for (int i = tid; i < DZ; i += 256) { c_benc[i] = benc[i]; c_wvar[i] = Wvar[i]; }
    for (int i = tid; i < DZ * NCLS; i += 256) c_wdec[i] = Wdec[i];
    if (tid < NCLS) c_bdec[tid] = bdec[tid];
    if (tid == 0)   c_bvar = bvar[0];

    // ---- wave / tile assignment ----
    const int lane = tid & 31;
    const int wv   = tid >> 5;
    const int rt   = wv & 3;          // 16-row tile 0..3
    const int ctb  = (wv >> 2) * 64;  // column base 0 or 64
    const int m16  = lane & 15;
    const int half = lane >> 4;

    v8f acc[4];
    #pragma unroll
    for (int j = 0; j < 4; ++j) acc[j] = (v8f){0.f,0.f,0.f,0.f,0.f,0.f,0.f,0.f};

    // prefetch registers for software pipelining
    float4 rx[4];        // x tile: 64 rows x 16 quads -> 4 quads / thread
    float  rw[8][4];     // W tile: 128 cols x 16 k-quads -> 8 groups x 4 / thread

    // ---------- staging helpers (register <- global) ----------
    auto loadX = [&](int kc) {
        #pragma unroll
        for (int g = 0; g < 4; ++g) {
            const int idx = tid + 256 * g;
            const int m = idx >> 4, q = idx & 15;
            int rg = rowBase + m; if (rg >= B) rg = B - 1;
            const int k0 = kc + 4 * q;
            float4 v = make_float4(0.f, 0.f, 0.f, 0.f);
            if (k0 + 3 < D_IN) {
                v = *(const float4*)(x + (size_t)rg * D_IN + k0);
            } else {
                if (k0 + 0 < D_IN) v.x = x[(size_t)rg * D_IN + k0 + 0];
                if (k0 + 1 < D_IN) v.y = x[(size_t)rg * D_IN + k0 + 1];
                if (k0 + 2 < D_IN) v.z = x[(size_t)rg * D_IN + k0 + 2];
                if (k0 + 3 < D_IN) v.w = x[(size_t)rg * D_IN + k0 + 3];
            }
            rx[g] = v;
        }
    };
    auto loadW = [&](int kc) {
        #pragma unroll
        for (int g = 0; g < 8; ++g) {
            const int idx = tid + 256 * g;
            const int n = idx & 127, kq = idx >> 7;   // wave reads 32 consecutive n: coalesced
            #pragma unroll
            for (int i = 0; i < 4; ++i) {
                const int k = kc + 4 * kq + i;
                rw[g][i] = (k < D_IN) ? Wenc[(size_t)k * DZ + n] : 0.f;
            }
        }
    };
    // ---------- staging helpers (LDS <- register, fp32 -> bf16 hi/lo split) ----------
    auto storeX = [&]() {
        #pragma unroll
        for (int g = 0; g < 4; ++g) {
            const int idx = tid + 256 * g;
            const int m = idx >> 4, q = idx & 15;
            const float f[4] = {rx[g].x, rx[g].y, rx[g].z, rx[g].w};
            H4 hi, lo;
            #pragma unroll
            for (int i = 0; i < 4; ++i) {
                const __bf16 h = (__bf16)f[i];
                hi.h[i] = h;
                lo.h[i] = (__bf16)(f[i] - (float)h);
            }
            *(uint2*)(sxh + m * SRS + 4 * q) = hi.u;
            *(uint2*)(sxl + m * SRS + 4 * q) = lo.u;
        }
    };
    auto storeW = [&]() {
        #pragma unroll
        for (int g = 0; g < 8; ++g) {
            const int idx = tid + 256 * g;
            const int n = idx & 127, kq = idx >> 7;
            H4 hi, lo;
            #pragma unroll
            for (int i = 0; i < 4; ++i) {
                const __bf16 h = (__bf16)rw[g][i];
                hi.h[i] = h;
                lo.h[i] = (__bf16)(rw[g][i] - (float)h);
            }
            *(uint2*)(swh + n * SRS + 4 * kq) = hi.u;
            *(uint2*)(swl + n * SRS + 4 * kq) = lo.u;
        }
    };

    // =================== GEMM: h = x @ W_enc, bf16x3 split via XDL WMMA ===================
    const __bf16* axh = sxh + (rt * 16 + m16) * SRS;
    const __bf16* axl = sxl + (rt * 16 + m16) * SRS;
    const __bf16* bwh = swh + (ctb + m16) * SRS;
    const __bf16* bwl = swl + (ctb + m16) * SRS;

    loadX(0); loadW(0);
    storeX(); storeW();
    __syncthreads();

    for (int c = 0; c < NCH; ++c) {
        const bool more = (c + 1 < NCH);
        if (more) { loadX((c + 1) * KC); loadW((c + 1) * KC); }  // overlap with compute

        #pragma unroll
        for (int ks = 0; ks < KC; ks += 32) {
            const int off = ks + 8 * half;
            const v16bf ah = ldfrag(axh, off);
            const v16bf al = ldfrag(axl, off);
            #pragma unroll
            for (int j = 0; j < 4; ++j) {
                const v16bf bh = ldfrag(bwh + j * 16 * SRS, off);
                const v16bf bl = ldfrag(bwl + j * 16 * SRS, off);
                acc[j] = __builtin_amdgcn_wmma_f32_16x16x32_bf16(false, ah, false, bh, (short)0, acc[j], false, false);
                acc[j] = __builtin_amdgcn_wmma_f32_16x16x32_bf16(false, ah, false, bl, (short)0, acc[j], false, false);
                acc[j] = __builtin_amdgcn_wmma_f32_16x16x32_bf16(false, al, false, bh, (short)0, acc[j], false, false);
            }
        }
        __syncthreads();
        if (more) { storeX(); storeW(); __syncthreads(); }
    }
    __syncthreads();

    // ---- park h (+ b_enc) in LDS: C-frag layout -> M = r + 8*half, N = m16 + 16j + ctb ----
    #pragma unroll
    for (int r = 0; r < 8; ++r) {
        const int mrow = rt * 16 + r + 8 * half;
        const int n0   = ctb + m16;
        sh[mrow * SHS + n0 +  0] = acc[0][r] + c_benc[n0 +  0];
        sh[mrow * SHS + n0 + 16] = acc[1][r] + c_benc[n0 + 16];
        sh[mrow * SHS + n0 + 32] = acc[2][r] + c_benc[n0 + 32];
        sh[mrow * SHS + n0 + 48] = acc[3][r] + c_benc[n0 + 48];
    }
    __syncthreads();

    // =================== per-row vMF epilogue: 4 lanes / row ===================
    const int rl   = tid >> 2;
    const int sub  = tid & 3;
    const int rowg = rowBase + rl;

    float hreg[32], xr[32];
    float ss = 0.f, hv = 0.f;
    #pragma unroll
    for (int i = 0; i < 32; ++i) {
        const int j = sub + 4 * i;
        const float h = sh[rl * SHS + j];
        hreg[i] = h;
        ss += h * h;
        hv += h * c_wvar[j];
    }
    ss = redg4(ss); hv = redg4(hv);
    const float rinv = 1.f / (sqrtf(ss) + 1e-8f);

    // kappa = softplus(h.Wvar + bvar) + 1
    const float pre   = hv + c_bvar;
    const float kappa = fmaxf(pre, 0.f) + log1pf(expf(-fabsf(pre))) + 1.f;

    // vMF envelope params (m = 128)
    const float mm1 = 127.f;
    const float cc  = sqrtf(4.f * kappa * kappa + mm1 * mm1);
    const float bt  = (cc - 2.f * kappa) / mm1;
    const float ba  = mm1 / (4.f * kappa);
    const float scl = fminf(fmaxf(kappa - 10.f, 0.f), 1.f);
    const float bb  = ba * scl + bt * (1.f - scl);
    const float aa  = (mm1 + 2.f * kappa + cc) * 0.25f;
    const float dd  = 4.f * aa * bb / (1.f + bb) - mm1 * 4.84418709f; // ln(127)

    // rejection sampler: first accepted of 64 proposals (fallback: last)
    const uint32_t sd = (uint32_t)rowg * 0x9E3779B9u + 0x7F4A7C15u;
    float w = 0.f; bool found = false;
    for (int k = 0; k < KPROP; ++k) {
        const float zn = gauss(hmix(sd ^ (uint32_t)(3 * k + 1) * 0x85EBCA77u),
                               hmix(sd ^ (uint32_t)(3 * k + 2) * 0x85EBCA77u));
        const float e  = fminf(fmaxf(0.5f + 0.04419417f * zn, 1e-6f), 1.f - 1e-6f);
        const float u  = u01(hmix(sd ^ (uint32_t)(3 * k + 3) * 0x85EBCA77u));
        const float den = 1.f - (1.f - bb) * e;
        const float wk  = (1.f - (1.f + bb) * e) / den;
        const float t   = 2.f * aa * bb / den;
        const bool  ok  = (mm1 * logf(t) - t + dd) > logf(u);
        if (!found) { w = wk; if (ok) found = true; }
    }
    const float wperp = sqrtf(fmaxf(1.f - w * w, 1e-10f));

    // tangent direction: 127-dim gaussian, normalized
    float vss = 0.f;
    #pragma unroll
    for (int i = 0; i < 32; ++i) {
        const int j = sub + 4 * i;
        float n = 0.f;
        if (j > 0) {
            n = gauss(hmix(sd ^ (uint32_t)(1000 + 2 * j) * 0xC2B2AE3Du),
                      hmix(sd ^ (uint32_t)(1001 + 2 * j) * 0xC2B2AE3Du));
        }
        xr[i] = n;
        vss += n * n;
    }
    vss = redg4(vss);
    const float vinv = 1.f / sqrtf(fmaxf(vss, 1e-20f));

    // mu, x_s (sample around e1), Householder vector uh = e1 - mu
    float uss = 0.f;
    #pragma unroll
    for (int i = 0; i < 32; ++i) {
        const int j  = sub + 4 * i;
        const float mu = hreg[i] * rinv;
        hreg[i] = mu;
        const float uh = (j == 0) ? (1.f - mu) : (-mu);
        uss += uh * uh;
        xr[i] = (j == 0) ? w : (wperp * xr[i] * vinv);
    }
    uss = redg4(uss);
    const float uinv = 1.f / (sqrtf(uss) + 1e-5f);

    float dxu = 0.f;
    #pragma unroll
    for (int i = 0; i < 32; ++i) {
        const int j = sub + 4 * i;
        const float uh = ((j == 0) ? (1.f - hreg[i]) : (-hreg[i])) * uinv;
        dxu += xr[i] * uh;
    }
    dxu = redg4(dxu);
    const float c2 = 2.f * dxu;

    // z = x_s - 2 (x_s . uh) uh, then decoder logits
    float dacc[NCLS];
    #pragma unroll
    for (int c = 0; c < NCLS; ++c) dacc[c] = 0.f;
    #pragma unroll
    for (int i = 0; i < 32; ++i) {
        const int j = sub + 4 * i;
        const float uh = ((j == 0) ? (1.f - hreg[i]) : (-hreg[i])) * uinv;
        const float zj = xr[i] - c2 * uh;
        #pragma unroll
        for (int c = 0; c < NCLS; ++c) dacc[c] += zj * c_wdec[j * NCLS + c];
    }
    #pragma unroll
    for (int c = 0; c < NCLS; ++c) dacc[c] = redg4(dacc[c]);
    if (sub == 0 && rowg < B) {
        #pragma unroll
        for (int c = 0; c < NCLS; ++c) out[(size_t)rowg * NCLS + c] = dacc[c] + c_bdec[c];
    }
}

extern "C" void kernel_launch(void* const* d_in, const int* in_sizes, int n_in,
                              void* d_out, int out_size, void* d_ws, size_t ws_size,
                              hipStream_t stream) {
    const float* x    = (const float*)d_in[0];
    const float* Wenc = (const float*)d_in[1];
    const float* benc = (const float*)d_in[2];
    const float* Wvar = (const float*)d_in[3];
    const float* bvar = (const float*)d_in[4];
    const float* Wdec = (const float*)d_in[5];
    const float* bdec = (const float*)d_in[6];
    float* out = (float*)d_out;

    const int B    = in_sizes[0] / D_IN;
    const int grid = (B + MT - 1) / MT;
    vib_vmf_kernel<<<grid, 256, 0, stream>>>(x, Wenc, benc, Wvar, bvar, Wdec, bdec, out, B);
}
